// BlockMoE_27195732918427
// MI455X (gfx1250) — compile-verified, matrix operand order checked by reading, not run on hardware
//
#include <hip/hip_runtime.h>
#include <hip/hip_bf16.h>
#include <math.h>

// ---------------- problem constants ----------------
constexpr int Bc = 2, Tc = 2048, Dc = 1024, Hc = 16, Ec = 8, Fc = 4096, HDc = 64;
constexpr int BT = Bc * Tc;           // 4096 tokens
constexpr int SLOTS = BT * 2;         // 8192 (token, k) slots, K=2
constexpr int CAP = SLOTS + Ec * 128; // padded capacity (segments padded to 128) -> 9216

// ---------------- types ----------------
typedef __bf16 bf16;
typedef __attribute__((ext_vector_type(16))) __bf16 bf16x16;
typedef __attribute__((ext_vector_type(8)))  float  v8f;

__device__ __forceinline__ v8f wmma_bf16(bf16x16 a, bf16x16 b, v8f c) {
  return __builtin_amdgcn_wmma_f32_16x16x32_bf16(
      /*neg_a=*/false, a, /*neg_b=*/false, b,
      /*c_mod=*/(short)0, c, /*reuse_a=*/false, /*reuse_b=*/false);
}

// 16-bit A/B fragment K-index map (ISA 7.12.2, 16-bit 16x32 A-matrix):
// lane-half 0 holds K {0..7, 16..23}; half 1 holds K {8..15, 24..31}
__device__ __forceinline__ int kmap(int i, int half) {
  return (i < 8) ? (half * 8 + i) : (8 + half * 8 + i);
}

__device__ __forceinline__ bf16 f2bf(float f) {
  union { float f; unsigned u; } c{f};
  unsigned r = (c.u + 0x7FFFu + ((c.u >> 16) & 1u)) >> 16;
  union { unsigned short s; bf16 b; } o{(unsigned short)r};
  return o.b;
}

__device__ __forceinline__ float geluf(float x) {
  return 0.5f * x * (1.0f + erff(x * 0.70710678118654752f));
}

// ---------------- CDNA5 async global->LDS copy (ASYNCcnt path) ----------------
// 16B per lane, written directly to LDS by the async engine.
__device__ __forceinline__ void async_b128(void* lds_dst, const void* gsrc) {
  const unsigned l = (unsigned)(uintptr_t)lds_dst; // low 32 bits = LDS offset
  asm volatile("global_load_async_to_lds_b128 %0, %1, off"
               :: "v"(l), "v"(gsrc) : "memory");
}
__device__ __forceinline__ void wait_async0() {
  asm volatile("s_wait_asynccnt 0" ::: "memory");
}

// ---------------- elementwise cast fp32 -> bf16 ----------------
__global__ __launch_bounds__(256) void cast_kernel(const float* __restrict__ in,
                                                   bf16* __restrict__ out, int n) {
  for (int i = blockIdx.x * 256 + threadIdx.x; i < n; i += gridDim.x * 256)
    out[i] = f2bf(in[i]);
}

// ---------------- LayerNorm (row of D=1024), writes f32 and/or bf16 ----------------
__global__ __launch_bounds__(256) void ln_kernel(const float* __restrict__ x,
                                                 const float* __restrict__ w,
                                                 float* __restrict__ outf,
                                                 bf16* __restrict__ outb) {
  __shared__ float sbuf[8];
  __shared__ float smu, srstd;
  const int row = blockIdx.x;
  const int tid = threadIdx.x;
  const float* xp = x + (size_t)row * Dc;
  float v[4], s = 0.f;
  for (int i = 0; i < 4; ++i) { v[i] = xp[tid + i * 256]; s += v[i]; }
  for (int off = 1; off < 32; off <<= 1) s += __shfl_xor(s, off, 32);
  if ((tid & 31) == 0) sbuf[tid >> 5] = s;
  __syncthreads();
  if (tid == 0) { float t = 0.f; for (int i = 0; i < 8; ++i) t += sbuf[i]; smu = t * (1.0f / Dc); }
  __syncthreads();
  const float mu = smu;
  float var = 0.f;
  for (int i = 0; i < 4; ++i) { float d = v[i] - mu; var += d * d; }
  for (int off = 1; off < 32; off <<= 1) var += __shfl_xor(var, off, 32);
  __syncthreads();
  if ((tid & 31) == 0) sbuf[tid >> 5] = var;
  __syncthreads();
  if (tid == 0) { float t = 0.f; for (int i = 0; i < 8; ++i) t += sbuf[i]; srstd = rsqrtf(t * (1.0f / Dc) + 1e-5f); }
  __syncthreads();
  const float rstd = srstd;
  for (int i = 0; i < 4; ++i) {
    int col = tid + i * 256;
    float val = (v[i] - mu) * rstd * w[col];
    if (outf) outf[(size_t)row * Dc + col] = val;
    if (outb) outb[(size_t)row * Dc + col] = f2bf(val);
  }
}

// ---------------- generic WMMA GEMM: 128x128x(K) tiles, 8 waves ----------------
// MODE 0: C(bf16) = A @ B                      (QKV)
// MODE 1: C(f32)  = A @ B + R                  (proj + residual)
// MODE 2: C(bf16) = gelu(gather(A,perm) @ B_e) (MoE up, grouped by expert segment)
// MODE 3: eo(f32)[slot] = topw[slot] * (A @ B_e)  scattered via perm   (MoE down)
template <int MODE>
__global__ __launch_bounds__(256) void gemm_kernel(
    const bf16* __restrict__ A, const bf16* __restrict__ Bw,
    void* __restrict__ Cout, const float* __restrict__ R,
    int M, int N, int Kd,
    const int* __restrict__ perm, const float* __restrict__ topw,
    const int* __restrict__ segoff) {
  const int mbase = blockIdx.y * 128;
  const int nbase = blockIdx.x * 128;
  const bf16* Bp = Bw;
  if (MODE == 2 || MODE == 3) {
    if (mbase >= segoff[8]) return;              // beyond padded total
    int e = 0;
    while (e < 7 && mbase >= segoff[e + 1]) ++e; // segments are 128-aligned
    Bp = Bw + (size_t)e * Kd * N;
  }

  __shared__ __align__(16) bf16 As[128 * 32];
  __shared__ __align__(16) bf16 Bs[32 * 128];

  const int tid = threadIdx.x;
  const int lane = tid & 31, wave = tid >> 5;
  const int wm = wave >> 2, wn = wave & 3;
  const int half = lane >> 4, l16 = lane & 15;

  const int aIdx = tid * 16, arow = aIdx >> 5, acol = aIdx & 31;
  const int bIdx = tid * 16, brow = bIdx >> 7, bcol = bIdx & 127;

  const v8f vz = {0.f, 0.f, 0.f, 0.f, 0.f, 0.f, 0.f, 0.f};
  v8f acc[4][2];
  for (int i = 0; i < 4; ++i) for (int j = 0; j < 2; ++j) acc[i][j] = vz;

  for (int k0 = 0; k0 < Kd; k0 += 32) {
    // stage A tile
    if (MODE == 2) {
      // gather with zero-fill for pad rows: synchronous path
      uint4 d0 = make_uint4(0, 0, 0, 0), d1 = d0;
      const long gr = perm[mbase + arow];
      if (gr >= 0) {
        const uint4* s = reinterpret_cast<const uint4*>(A + (size_t)gr * Kd + k0 + acol);
        d0 = s[0]; d1 = s[1];
      }
      uint4* dst = reinterpret_cast<uint4*>(&As[arow * 32 + acol]);
      dst[0] = d0; dst[1] = d1;
    } else {
      const bf16* gA = A + (size_t)(mbase + arow) * Kd + k0 + acol;
      async_b128(&As[arow * 32 + acol],      gA);
      async_b128(&As[arow * 32 + acol + 8],  gA + 8);
    }
    // stage B tile (async engine writes LDS directly)
    {
      const bf16* gB = Bp + (size_t)(k0 + brow) * N + nbase + bcol;
      async_b128(&Bs[brow * 128 + bcol],     gB);
      async_b128(&Bs[brow * 128 + bcol + 8], gB + 8);
    }
    if (k0 + 32 < Kd) {
      __builtin_prefetch(Bp + (size_t)(k0 + 32 + brow) * N + nbase + bcol, 0, 1);
      if (MODE != 2)
        __builtin_prefetch(A + (size_t)(mbase + arow) * Kd + k0 + 32 + acol, 0, 1);
    }
    wait_async0();
    __syncthreads();

    bf16x16 af[4], bfr[2];
    for (int sm = 0; sm < 4; ++sm) {
      const int r = wm * 64 + sm * 16 + l16;
      for (int i = 0; i < 16; ++i) af[sm][i] = As[r * 32 + kmap(i, half)];
    }
    for (int sn = 0; sn < 2; ++sn) {
      const int c = wn * 32 + sn * 16 + l16;
      for (int i = 0; i < 16; ++i) bfr[sn][i] = Bs[kmap(i, half) * 128 + c];
    }
    for (int sm = 0; sm < 4; ++sm)
      for (int sn = 0; sn < 2; ++sn)
        acc[sm][sn] = wmma_bf16(af[sm], bfr[sn], acc[sm][sn]);
    __syncthreads();
  }

  // epilogue (C-matrix layout: lane half selects M+8, l16 selects N)
  for (int sm = 0; sm < 4; ++sm)
    for (int sn = 0; sn < 2; ++sn)
      for (int r = 0; r < 8; ++r) {
        const int row = mbase + wm * 64 + sm * 16 + r + 8 * half;
        const int col = nbase + wn * 32 + sn * 16 + l16;
        const float v = acc[sm][sn][r];
        if (MODE == 0) {
          ((bf16*)Cout)[(size_t)row * N + col] = f2bf(v);
        } else if (MODE == 1) {
          const size_t o = (size_t)row * N + col;
          ((float*)Cout)[o] = v + R[o];
        } else if (MODE == 2) {
          ((bf16*)Cout)[(size_t)row * N + col] = f2bf(geluf(v));
        } else {
          const int slot = perm[row];
          if (slot >= 0) ((float*)Cout)[(size_t)slot * N + col] = topw[slot] * v;
        }
      }
}

// ---------------- flash attention: 64-q tile per block (4 waves), WMMA both GEMMs ----------------
__global__ __launch_bounds__(128) void attn_kernel(const bf16* __restrict__ qkv,
                                                   bf16* __restrict__ out) {
  __shared__ __align__(16) bf16 Ks[64 * 64];
  __shared__ __align__(16) bf16 Vs[64 * 64];
  __shared__ __align__(16) bf16 Ps[4][16 * 64];

  const int bh = blockIdx.y;            // 0..31
  const int b = bh >> 4, h = bh & 15;
  const int qt = blockIdx.x;            // 0..T/64-1
  const int tid = threadIdx.x;
  const int wave = tid >> 5, lane = tid & 31;
  const int half = lane >> 4, l16 = lane & 15;
  const int q0 = qt * 64 + wave * 16;
  const int RS = 3 * Dc;                // 3072 row stride in qkv
  const bf16* base = qkv + (size_t)b * Tc * RS;

  // Q fragments (A-layout), HD=64 -> two 32-wide k-steps
  bf16x16 qf[2];
  {
    const bf16* qp = base + (size_t)(q0 + l16) * RS + h * HDc;
    for (int ks = 0; ks < 2; ++ks)
      for (int i = 0; i < 16; ++i) qf[ks][i] = qp[ks * 32 + kmap(i, half)];
  }

  const v8f vz = {0.f, 0.f, 0.f, 0.f, 0.f, 0.f, 0.f, 0.f};
  v8f o[4];
  for (int dt = 0; dt < 4; ++dt) o[dt] = vz;
  float mrow[8], lrow[8];
  for (int r = 0; r < 8; ++r) { mrow[r] = -1e30f; lrow[r] = 0.f; }

  for (int kt = 0; kt <= qt; ++kt) {
    // async K/V tile stage (64x64 bf16 each)
    for (int j = 0; j < 2; ++j) {
      const int idx = (tid + j * 128) * 16;
      const int kr = idx >> 6, kc = idx & 63;
      const bf16* kp = base + (size_t)(kt * 64 + kr) * RS + Dc + h * HDc + kc;
      const bf16* vp = kp + Dc;
      async_b128(&Ks[kr * 64 + kc],     kp);
      async_b128(&Ks[kr * 64 + kc + 8], kp + 8);
      async_b128(&Vs[kr * 64 + kc],     vp);
      async_b128(&Vs[kr * 64 + kc + 8], vp + 8);
    }
    wait_async0();
    __syncthreads();

    // S = Q @ K^T (4 n-tiles of 16 keys)
    v8f s[4];
    for (int nt = 0; nt < 4; ++nt) {
      s[nt] = vz;
      for (int ks = 0; ks < 2; ++ks) {
        bf16x16 kf;
        const int col = nt * 16 + l16;
        for (int i = 0; i < 16; ++i) kf[i] = Ks[col * 64 + ks * 32 + kmap(i, half)];
        s[nt] = wmma_bf16(qf[ks], kf, s[nt]);
      }
    }
    // scale + causal mask
    for (int nt = 0; nt < 4; ++nt)
      for (int r = 0; r < 8; ++r) {
        const int key = kt * 64 + nt * 16 + l16;
        const int qr = q0 + r + 8 * half;
        const float v = s[nt][r] * 0.125f;   // 1/sqrt(64)
        s[nt][r] = (key <= qr) ? v : -1e30f;
      }
    // online softmax per row (rows live in a 16-lane half)
    for (int r = 0; r < 8; ++r) {
      float tm = s[0][r];
      for (int nt = 1; nt < 4; ++nt) tm = fmaxf(tm, s[nt][r]);
      for (int off = 1; off < 16; off <<= 1) tm = fmaxf(tm, __shfl_xor(tm, off, 32));
      const float mnew = fmaxf(mrow[r], tm);
      const float alpha = __expf(mrow[r] - mnew);
      float ps = 0.f;
      for (int nt = 0; nt < 4; ++nt) {
        const float p = __expf(s[nt][r] - mnew);
        s[nt][r] = p;
        ps += p;
      }
      for (int off = 1; off < 16; off <<= 1) ps += __shfl_xor(ps, off, 32);
      lrow[r] = lrow[r] * alpha + ps;
      mrow[r] = mnew;
      for (int dt = 0; dt < 4; ++dt) o[dt][r] *= alpha;
    }
    // relay P through per-wave LDS: C-layout -> A-layout
    for (int nt = 0; nt < 4; ++nt)
      for (int r = 0; r < 8; ++r)
        Ps[wave][(r + 8 * half) * 64 + nt * 16 + l16] = f2bf(s[nt][r]);
    asm volatile("s_wait_dscnt 0" ::: "memory");
    // O += P @ V
    for (int ks = 0; ks < 2; ++ks) {
      bf16x16 pf;
      for (int i = 0; i < 16; ++i) pf[i] = Ps[wave][l16 * 64 + ks * 32 + kmap(i, half)];
      for (int dt = 0; dt < 4; ++dt) {
        bf16x16 vf;
        for (int i = 0; i < 16; ++i) vf[i] = Vs[(ks * 32 + kmap(i, half)) * 64 + dt * 16 + l16];
        o[dt] = wmma_bf16(pf, vf, o[dt]);
      }
    }
    __syncthreads();
  }

  for (int dt = 0; dt < 4; ++dt)
    for (int r = 0; r < 8; ++r) {
      const int qr = q0 + r + 8 * half;
      const float v = o[dt][r] / lrow[r];
      out[(size_t)(b * Tc + qr) * Dc + h * HDc + dt * 16 + l16] = f2bf(v);
    }
}

// ---------------- router: one wave per token, softmax + top-2 ----------------
__global__ __launch_bounds__(128) void router_kernel(const float* __restrict__ xln,
                                                     const float* __restrict__ rw,
                                                     int* __restrict__ topi,
                                                     float* __restrict__ topw) {
  const int wave = threadIdx.x >> 5, lane = threadIdx.x & 31;
  const int t = blockIdx.x * 4 + wave;
  const float* xp = xln + (size_t)t * Dc;
  float acc[Ec];
  for (int e = 0; e < Ec; ++e) acc[e] = 0.f;
  for (int d = lane; d < Dc; d += 32) {
    const float xv = xp[d];
    for (int e = 0; e < Ec; ++e) acc[e] += xv * rw[d * Ec + e];
  }
  for (int e = 0; e < Ec; ++e)
    for (int off = 1; off < 32; off <<= 1) acc[e] += __shfl_xor(acc[e], off, 32);
  if (lane == 0) {
    float mx = acc[0];
    for (int e = 1; e < Ec; ++e) mx = fmaxf(mx, acc[e]);
    float p[Ec];
    for (int e = 0; e < Ec; ++e) p[e] = __expf(acc[e] - mx);
    int e0 = 0;
    for (int e = 1; e < Ec; ++e) if (p[e] > p[e0]) e0 = e;
    int e1 = (e0 == 0) ? 1 : 0;
    for (int e = 0; e < Ec; ++e) if (e != e0 && p[e] > p[e1]) e1 = e;
    const float s = p[e0] + p[e1];
    topi[2 * t] = e0;     topi[2 * t + 1] = e1;
    topw[2 * t] = p[e0] / s; topw[2 * t + 1] = p[e1] / s;
  }
}

// ---------------- expert segment offsets (padded to 128) ----------------
__global__ __launch_bounds__(256) void offsets_kernel(const int* __restrict__ topi,
                                                      int* __restrict__ off,
                                                      int* __restrict__ cursor) {
  __shared__ int cnt[Ec];
  const int tid = threadIdx.x;
  if (tid < Ec) cnt[tid] = 0;
  __syncthreads();
  for (int s = tid; s < SLOTS; s += 256) atomicAdd(&cnt[topi[s]], 1);
  __syncthreads();
  if (tid == 0) {
    int acc = 0;
    for (int e = 0; e < Ec; ++e) { off[e] = acc; acc += (cnt[e] + 127) & ~127; }
    off[8] = acc;
  }
  if (tid < Ec) cursor[tid] = 0;
}

__global__ __launch_bounds__(256) void perm_init_kernel(int* __restrict__ perm) {
  const int i = blockIdx.x * 256 + threadIdx.x;
  if (i < CAP) perm[i] = -1;
}

__global__ __launch_bounds__(256) void scatter_kernel(const int* __restrict__ topi,
                                                      const int* __restrict__ off,
                                                      int* __restrict__ cursor,
                                                      int* __restrict__ perm) {
  const int s = blockIdx.x * 256 + threadIdx.x;
  if (s < SLOTS) {
    const int e = topi[s];
    const int pos = off[e] + atomicAdd(&cursor[e], 1);
    perm[pos] = s;
  }
}

// ---------------- final combine: out = xln + eo[2t] + eo[2t+1] ----------------
__global__ __launch_bounds__(256) void combine_kernel(const float* __restrict__ xln,
                                                      const float* __restrict__ eo,
                                                      float* __restrict__ out) {
  const size_t i = (size_t)blockIdx.x * 256 + threadIdx.x;
  const int t = (int)(i >> 10), d = (int)(i & 1023);
  out[i] = xln[i] + eo[(size_t)(2 * t) * Dc + d] + eo[(size_t)(2 * t + 1) * Dc + d];
}

// ---------------- host launch ----------------
extern "C" void kernel_launch(void* const* d_in, const int* in_sizes, int n_in,
                              void* d_out, int out_size, void* d_ws, size_t ws_size,
                              hipStream_t stream) {
  (void)in_sizes; (void)n_in; (void)out_size; (void)ws_size;
  const float* x     = (const float*)d_in[0];
  const float* ln1w  = (const float*)d_in[1];
  const float* wqkv  = (const float*)d_in[2];
  const float* wproj = (const float*)d_in[3];
  const float* ln2w  = (const float*)d_in[4];
  const float* rw    = (const float*)d_in[5];
  const float* w1    = (const float*)d_in[6];
  const float* w2    = (const float*)d_in[7];
  float* out = (float*)d_out;

  // workspace carve-up
  char* p = (char*)d_ws;
  size_t cur = 0;
  auto alloc = [&](size_t bytes) { void* r = p + cur; cur += (bytes + 255) & ~(size_t)255; return r; };
  bf16*  h_bf     = (bf16*) alloc((size_t)BT * Dc * 2);
  bf16*  wqkv_bf  = (bf16*) alloc((size_t)Dc * 3 * Dc * 2);
  bf16*  wproj_bf = (bf16*) alloc((size_t)Dc * Dc * 2);
  bf16*  w1_bf    = (bf16*) alloc((size_t)Ec * Dc * Fc * 2);
  bf16*  w2_bf    = (bf16*) alloc((size_t)Ec * Fc * Dc * 2);
  bf16*  qkv_bf   = (bf16*) alloc((size_t)BT * 3 * Dc * 2);
  bf16*  attn_bf  = (bf16*) alloc((size_t)BT * Dc * 2);
  float* xres     = (float*)alloc((size_t)BT * Dc * 4);
  float* xlnf     = (float*)alloc((size_t)BT * Dc * 4);
  bf16*  xlnb     = (bf16*) alloc((size_t)BT * Dc * 2);
  int*   topi     = (int*)  alloc((size_t)SLOTS * 4);
  float* topw     = (float*)alloc((size_t)SLOTS * 4);
  int*   segoff   = (int*)  alloc(16 * 4);
  int*   cursor   = (int*)  alloc(16 * 4);
  int*   perm     = (int*)  alloc((size_t)CAP * 4);
  bf16*  hh       = (bf16*) alloc((size_t)CAP * Fc * 2);
  float* eo       = (float*)alloc((size_t)SLOTS * Dc * 4);

  // 1) weights -> bf16
  cast_kernel<<<2048, 256, 0, stream>>>(wqkv,  wqkv_bf,  Dc * 3 * Dc);
  cast_kernel<<<2048, 256, 0, stream>>>(wproj, wproj_bf, Dc * Dc);
  cast_kernel<<<8192, 256, 0, stream>>>(w1,    w1_bf,    Ec * Dc * Fc);
  cast_kernel<<<8192, 256, 0, stream>>>(w2,    w2_bf,    Ec * Fc * Dc);

  // 2) LN1 -> bf16 activations
  ln_kernel<<<BT, 256, 0, stream>>>(x, ln1w, nullptr, h_bf);

  // 3) QKV = h @ wqkv  (4096 x 3072 x 1024), bf16 out
  gemm_kernel<0><<<dim3(3 * Dc / 128, BT / 128), 256, 0, stream>>>(
      h_bf, wqkv_bf, qkv_bf, nullptr, BT, 3 * Dc, Dc, nullptr, nullptr, nullptr);

  // 4) flash attention
  attn_kernel<<<dim3(Tc / 64, Bc * Hc), 128, 0, stream>>>(qkv_bf, attn_bf);

  // 5) xres = x + attn @ wproj
  gemm_kernel<1><<<dim3(Dc / 128, BT / 128), 256, 0, stream>>>(
      attn_bf, wproj_bf, xres, x, BT, Dc, Dc, nullptr, nullptr, nullptr);

  // 6) LN2 -> f32 (residual base) + bf16 (GEMM input)
  ln_kernel<<<BT, 256, 0, stream>>>(xres, ln2w, xlnf, xlnb);

  // 7) router top-2
  router_kernel<<<BT / 4, 128, 0, stream>>>(xlnf, rw, topi, topw);

  // 8) build 128-padded expert segments + permutation
  offsets_kernel<<<1, 256, 0, stream>>>(topi, segoff, cursor);
  perm_init_kernel<<<(CAP + 255) / 256, 256, 0, stream>>>(perm);
  scatter_kernel<<<SLOTS / 256, 256, 0, stream>>>(topi, segoff, cursor, perm);

  // 9) MoE up: hh = gelu(gather(xln) @ w1[e])   (CAP x 4096 x 1024)
  gemm_kernel<2><<<dim3(Fc / 128, CAP / 128), 256, 0, stream>>>(
      xlnb, w1_bf, hh, nullptr, CAP, Fc, Dc, perm, nullptr, segoff);

  // 10) MoE down: eo[slot] = topw[slot] * (hh @ w2[e])   (CAP x 1024 x 4096)
  gemm_kernel<3><<<dim3(Dc / 128, CAP / 128), 256, 0, stream>>>(
      hh, w2_bf, eo, nullptr, CAP, Dc, Fc, perm, topw, segoff);

  // 11) out = xln + eo[2t] + eo[2t+1]
  combine_kernel<<<(BT * Dc) / 256, 256, 0, stream>>>(xlnf, eo, out);
}
